// Sae_45406394253320
// MI455X (gfx1250) — matrix operand order repaired and checked by reading, not compiled, not run on hardware
//
#include <hip/hip_runtime.h>
#include <hip/hip_bf16.h>
#include <stdint.h>

// ---------------------------------------------------------------------------
// SAE forward for MI455X (gfx1250): bf16 WMMA GEMMs + fused top-k histogram
//   encoder:  h = relu(x @ W_enc^T + b_enc)       (M=2048,N=16384,K=1024) x2 inst
//   top-k:    histogram (fused in encoder epilogue) -> threshold -> zero below
//   decoder:  x' = relu(h @ W_dec^T + b_dec)      (M=2048,N=1024,K=16384) x2 inst
// Block tile 256x128, BK=32, 8 waves, wave tile 64x64 (4x4 of 16x16 WMMA).
// ---------------------------------------------------------------------------

typedef __attribute__((ext_vector_type(16))) __bf16 v16bf;  // 8 VGPRs
typedef __attribute__((ext_vector_type(8)))  __bf16 v8bf;   // 4 VGPRs
typedef __attribute__((ext_vector_type(8)))  float  v8f;    // 8 VGPRs

static constexpr int  BATCH  = 2048;
static constexpr int  INST   = 2;
static constexpr int  IN_AE  = 1024;
static constexpr int  H_AE   = 16384;
static constexpr int  TOPK   = 64;
static constexpr long KEEP   = (long)BATCH * INST * TOPK;       // 262144
static constexpr long XOUT_ELEMS = (long)BATCH * INST * IN_AE;  // 4194304
static constexpr long H_ELEMS    = (long)BATCH * INST * H_AE;   // 67108864
static constexpr int  NBINS  = 2048;  // positive-float bits >> 20

__device__ __forceinline__ v8bf pack_bf16(float4 a, float4 b) {
    v8bf r;
    r[0] = (__bf16)a.x; r[1] = (__bf16)a.y; r[2] = (__bf16)a.z; r[3] = (__bf16)a.w;
    r[4] = (__bf16)b.x; r[5] = (__bf16)b.y; r[6] = (__bf16)b.z; r[7] = (__bf16)b.w;
    return r;
}

// C[m,n] = relu(sum_k A[m,k]*B[n,k] + bias[n]);  A:[M,K] rows lda, B:[N,K] rows ldb.
// grid: (M/256, N/128, INST)  -- M-tiles fastest for L2 weight residency.
template <bool DO_HIST>
__global__ __launch_bounds__(256)
void gemm_bias_relu(const float* __restrict__ A, long lda, long instA,
                    const float* __restrict__ B, long ldb, long instB,
                    const float* __restrict__ bias, long instBias,
                    float* __restrict__ C, long ldc, long instC,
                    int K, unsigned* __restrict__ gHist) {
    __shared__ __align__(16) __bf16 Al[256 * 32];   // 16 KB
    __shared__ __align__(16) __bf16 Bl[128 * 32];   // 8 KB
    __shared__ unsigned hs[NBINS];                  // 8 KB (used when DO_HIST)
    __shared__ __align__(16) float biasLds[128];    // async-loaded bias slice

    const int tid  = threadIdx.x;
    const int mT   = blockIdx.x;   // fastest-varying: consecutive blocks share B
    const int nT   = blockIdx.y;
    const int inst = blockIdx.z;

    const float* Ab = A + instA * inst + (long)mT * 256 * lda;
    const float* Bb = B + instB * inst + (long)nT * 128 * ldb;

    // ---- async bias load (ASYNCcnt path): overlaps the entire K loop ----
    if (tid < 128) {
        const float* gbias = bias + instBias * inst + (long)nT * 128 + tid;
        unsigned ldsOff = (unsigned)(unsigned long long)&biasLds[tid];
        asm volatile("global_load_async_to_lds_b32 %0, %1, off"
                     :: "v"(ldsOff), "v"(gbias) : "memory");
    }

    if (DO_HIST) {
        for (int b = tid; b < NBINS; b += 256) hs[b] = 0u;
    }

    // B staging: thread t -> row t/2, 16-element K slab at (t&1)*16
    const int srow = tid >> 1;
    const int scol = (tid & 1) * 16;

    const int lane = tid & 31;
    const int wave = tid >> 5;
    const int wm   = wave >> 1;      // 0..3 -> rows wm*64
    const int wn   = wave & 1;       // 0..1 -> cols wn*64
    const int lr   = lane & 15;
    const int hi   = lane >> 4;      // 0/1 half-wave

    v8f acc[4][4] = {};

    union Frag { uint4 q[2]; v16bf v; };

    for (int k0 = 0; k0 < K; k0 += 32) {
        // ---- stage A (256x32) and B (128x32): fp32 global -> bf16 LDS ----
        const float* ga = Ab + (long)tid * lda + k0;   // one full row per thread
        const float* gb = Bb + (long)srow * ldb + k0 + scol;
        float4 a0 = *(const float4*)(ga + 0);
        float4 a1 = *(const float4*)(ga + 4);
        float4 a2 = *(const float4*)(ga + 8);
        float4 a3 = *(const float4*)(ga + 12);
        float4 a4 = *(const float4*)(ga + 16);
        float4 a5 = *(const float4*)(ga + 20);
        float4 a6 = *(const float4*)(ga + 24);
        float4 a7 = *(const float4*)(ga + 28);
        float4 b0 = *(const float4*)(gb + 0);
        float4 b1 = *(const float4*)(gb + 4);
        float4 b2 = *(const float4*)(gb + 8);
        float4 b3 = *(const float4*)(gb + 12);
        if (k0 + 32 < K) {  // prefetch next K tile (global_prefetch_b8)
            __builtin_prefetch(ga + 32, 0, 1);
            __builtin_prefetch(gb + 32, 0, 1);
        }
        *(v8bf*)&Al[tid * 32 + 0]  = pack_bf16(a0, a1);
        *(v8bf*)&Al[tid * 32 + 8]  = pack_bf16(a2, a3);
        *(v8bf*)&Al[tid * 32 + 16] = pack_bf16(a4, a5);
        *(v8bf*)&Al[tid * 32 + 24] = pack_bf16(a6, a7);
        *(v8bf*)&Bl[srow * 32 + scol + 0] = pack_bf16(b0, b1);
        *(v8bf*)&Bl[srow * 32 + scol + 8] = pack_bf16(b2, b3);
        __syncthreads();

        // ---- load fragments per ISA 16-bit layouts ----
        Frag af[4];
#pragma unroll
        for (int sm = 0; sm < 4; ++sm) {
            const int row = wm * 64 + sm * 16 + lr;
            const int kb  = hi * 8;  // lanes 0-15: K0..7/K16..23 ; 16-31: K8..15/K24..31
            af[sm].q[0] = *(const uint4*)&Al[row * 32 + kb];
            af[sm].q[1] = *(const uint4*)&Al[row * 32 + kb + 16];
        }
        Frag bf[4];
#pragma unroll
        for (int sn = 0; sn < 4; ++sn) {
            const int col = wn * 64 + sn * 16 + lr;
            const int kb  = hi * 8;  // lanes 0-15: K0..7/K16..23 ; 16-31: K8..15/K24..31
            bf[sn].q[0] = *(const uint4*)&Bl[col * 32 + kb];
            bf[sn].q[1] = *(const uint4*)&Bl[col * 32 + kb + 16];
        }

#pragma unroll
        for (int sm = 0; sm < 4; ++sm)
#pragma unroll
            for (int sn = 0; sn < 4; ++sn)
                acc[sm][sn] = __builtin_amdgcn_wmma_f32_16x16x32_bf16(
                    /*neg_a=*/false, af[sm].v, /*neg_b=*/false, bf[sn].v,
                    /*c_mod=*/(short)0, acc[sm][sn],
                    /*reuse_a=*/false, /*reuse_b=*/false);
        __syncthreads();
    }

    // ---- bias must have landed in LDS by now ----
    asm volatile("s_wait_asynccnt 0x0" ::: "memory");
    __syncthreads();

    // ---- epilogue: bias + relu + store (+ LDS histogram) ----
    float* Cb = C + instC * inst;
#pragma unroll
    for (int sm = 0; sm < 4; ++sm) {
#pragma unroll
        for (int sn = 0; sn < 4; ++sn) {
            const int colL = wn * 64 + sn * 16 + lr;
            const int colN = nT * 128 + colL;
            const float bv = biasLds[colL];
#pragma unroll
            for (int r = 0; r < 8; ++r) {
                const int rowM = mT * 256 + wm * 64 + sm * 16 + r + hi * 8;
                float v = acc[sm][sn][r] + bv;
                v = fmaxf(v, 0.0f);
                Cb[(long)rowM * ldc + colN] = v;
                if (DO_HIST) {
                    if (v > 0.0f) {
                        const unsigned bin = __float_as_uint(v) >> 20;  // < 2048
                        atomicAdd(&hs[bin], 1u);
                    }
                }
            }
        }
    }
    if (DO_HIST) {
        __syncthreads();
        for (int b = tid; b < NBINS; b += 256) {
            const unsigned c = hs[b];
            if (c) atomicAdd(&gHist[b], c);
        }
    }
}

__global__ void zero_hist(unsigned* __restrict__ hist) {
    for (int i = threadIdx.x; i < NBINS; i += blockDim.x) hist[i] = 0u;
}

// descending scan over histogram bins; threshold = lower edge of crossing bin
__global__ void scan_threshold(const unsigned* __restrict__ hist,
                               float* __restrict__ thr) {
    if (threadIdx.x == 0 && blockIdx.x == 0) {
        unsigned long long cum = 0;
        float t = 0.0f;
        for (int b = NBINS - 1; b >= 0; --b) {
            cum += hist[b];
            if (cum >= (unsigned long long)KEEP) {
                t = __uint_as_float((unsigned)b << 20);
                break;
            }
        }
        *thr = t;
    }
}

__global__ __launch_bounds__(256)
void apply_threshold(float* __restrict__ h, const float* __restrict__ thrP, long n4) {
    const float t = *thrP;
    float4* h4 = (float4*)h;
    long i = (long)blockIdx.x * blockDim.x + threadIdx.x;
    const long stride = (long)gridDim.x * blockDim.x;
    for (; i < n4; i += stride) {
        float4 v = h4[i];
        v.x = (v.x >= t) ? v.x : 0.0f;
        v.y = (v.y >= t) ? v.y : 0.0f;
        v.z = (v.z >= t) ? v.z : 0.0f;
        v.w = (v.w >= t) ? v.w : 0.0f;
        h4[i] = v;
    }
}

extern "C" void kernel_launch(void* const* d_in, const int* in_sizes, int n_in,
                              void* d_out, int out_size, void* d_ws, size_t ws_size,
                              hipStream_t stream) {
    const float* x     = (const float*)d_in[0];  // [2048,2,1,1024]
    const float* W_enc = (const float*)d_in[1];  // [2,16384,1024]
    const float* W_dec = (const float*)d_in[2];  // [2,1024,16384]
    const float* b_enc = (const float*)d_in[3];  // [2,16384]
    const float* b_dec = (const float*)d_in[4];  // [2,1024]

    float* out    = (float*)d_out;
    float* xprime = out;               // [2048,2,1024]
    float* h      = out + XOUT_ELEMS;  // [2048,2,16384]

    unsigned* hist = (unsigned*)d_ws;
    float*    thr  = (float*)((char*)d_ws + NBINS * sizeof(unsigned));

    zero_hist<<<1, 256, 0, stream>>>(hist);

    // Encoder: M=2048, N=16384, K=1024.
    dim3 ge(BATCH / 256, H_AE / 128, INST);  // (8,128,2), mT fastest
    gemm_bias_relu<true><<<ge, 256, 0, stream>>>(
        x,     (long)INST * IN_AE,   (long)IN_AE,
        W_enc, (long)IN_AE,          (long)H_AE * IN_AE,
        b_enc, (long)H_AE,
        h,     (long)INST * H_AE,    (long)H_AE,
        IN_AE, hist);

    scan_threshold<<<1, 1, 0, stream>>>(hist, thr);
    apply_threshold<<<4096, 256, 0, stream>>>(h, thr, H_ELEMS / 4);

    // Decoder: M=2048, N=1024, K=16384.
    dim3 gd(BATCH / 256, IN_AE / 128, INST);  // (8,8,2)
    gemm_bias_relu<false><<<gd, 256, 0, stream>>>(
        h,     (long)INST * H_AE,    (long)H_AE,
        W_dec, (long)H_AE,           (long)IN_AE * H_AE,
        b_dec, (long)IN_AE,
        xprime,(long)INST * IN_AE,   (long)IN_AE,
        H_AE, nullptr);
}
